// BiLSTM_4389456576645
// MI455X (gfx1250) — matrix operand order repaired and checked by reading, not compiled, not run on hardware
//
#include <hip/hip_runtime.h>

// ---------------------------------------------------------------------------
// BiLSTM text classifier for MI455X (gfx1250, wave32, WMMA).
// B=128, T=512, V=50000, D=128, H=256, C=10
//  - weights (stacked [W;U]) packed once to bf16 in WMMA B-tile layout
//  - embedding lookup -> time-major bf16 [T,B,D]
//  - per timestep: fused GEMM z = [x_t ; h] @ [W;U] + bias via
//    v_wmma_f32_16x16x32_bf16 (f32 accum), activation fused in epilogue,
//    both directions per launch; K1 is a template param so the x/h phases
//    have compile-time trip counts (affine addressing, no per-iter selects).
//    Each wave owns TWO adjacent N-tiles sharing one A tile (A reuse + two
//    independent WMMA accumulator chains for load/MMA overlap).
//  - layer1 writes seq bf16 [T,B,2H]; layer2 keeps last states; dense+softmax.
// ---------------------------------------------------------------------------

typedef __bf16 bf16_t;
typedef __bf16 v16bf __attribute__((ext_vector_type(16)));
typedef __bf16 v8bf  __attribute__((ext_vector_type(8)));
typedef float  v8f   __attribute__((ext_vector_type(8)));

#define Bn   128
#define Tn   512
#define Dn   128
#define Hn   256
#define Cn   10
#define G4H  1024   // 4*H

__device__ __forceinline__ float fast_sigmoid(float x) {
    return 1.0f / (1.0f + __expf(-x));
}
__device__ __forceinline__ float fast_tanh(float x) {
    float e = __expf(2.0f * x);
    return (e - 1.0f) / (e + 1.0f);
}

// ---------------------------------------------------------------------------
// Zero helper (workspace state init; deterministic per call)
// ---------------------------------------------------------------------------
__global__ void k_zero_u32(unsigned int* p, int n) {
    int i = blockIdx.x * blockDim.x + threadIdx.x;
    if (i < n) p[i] = 0u;
}

// ---------------------------------------------------------------------------
// Pack stacked f32 weights [top(Ktop,1024) ; bot(Kbot,1024)] into bf16 WMMA
// B-matrix tile layout: element ((nt*KT + kt)*32 + lane)*16 + e maps to
//   n = nt*16 + (lane&15),  k = kt*32 + ((lane&16)?8:0) + (e&7) + ((e&8)?16:0)
// (v_wmma_f32_16x16x32_bf16 B operand: lanes 0-15 hold K=0..7,16..23,
//  lanes 16-31 hold K=8..15,24..31, 2 K-values packed per VGPR)
// ---------------------------------------------------------------------------
__global__ void k_pack_b(const float* __restrict__ top, int Ktop,
                         const float* __restrict__ bot, int Kbot,
                         bf16_t* __restrict__ out) {
    int K  = Ktop + Kbot;
    int KT = K >> 5;
    int total = K * G4H;
    int idx = blockIdx.x * blockDim.x + threadIdx.x;
    if (idx >= total) return;
    int e    = idx & 15;
    int lane = (idx >> 4) & 31;
    int tile = idx >> 9;
    int kt   = tile % KT;
    int nt   = tile / KT;
    int n = nt * 16 + (lane & 15);
    int k = kt * 32 + ((lane & 16) ? 8 : 0) + (e & 7) + ((e & 8) ? 16 : 0);
    float v = (k < Ktop) ? top[(size_t)k * G4H + n]
                         : bot[(size_t)(k - Ktop) * G4H + n];
    out[idx] = (bf16_t)v;
}

// ---------------------------------------------------------------------------
// Embedding lookup, f32 -> bf16, time-major: e[t][b][d] = emb[x[b][t]][d]
// ---------------------------------------------------------------------------
__global__ void k_embed(const int* __restrict__ x, const float* __restrict__ emb,
                        bf16_t* __restrict__ e_bf) {
    int idx = blockIdx.x * blockDim.x + threadIdx.x;   // Tn*Bn*Dn = 8.4M
    if (idx >= Tn * Bn * Dn) return;
    int d  = idx & (Dn - 1);
    int tb = idx >> 7;
    int b  = tb & (Bn - 1);
    int t  = tb >> 7;
    int tok = x[b * Tn + t];
    e_bf[idx] = (bf16_t)emb[(size_t)tok * Dn + d];
}

// ---------------------------------------------------------------------------
// Epilogue helper: bias + gate activation + coalesced store of one 16x16 tile
// C layout: acc[r] at row mt*16 + r + ((lane&16)?8:0), col nt*16 + (lane&15)
// ---------------------------------------------------------------------------
__device__ __forceinline__ void store_gate_tile(
        v8f acc, int mt, int nt, int lane,
        const float* __restrict__ bias, float* __restrict__ gout) {
    const int ncol    = nt * 16 + (lane & 15);
    const int gid     = nt >> 4;                  // 0:i 1:f 2:g 3:o
    const int rowbase = mt * 16 + ((lane & 16) ? 8 : 0);
    const float bv    = bias[ncol];
    #pragma unroll
    for (int r = 0; r < 8; ++r) {
        float z = acc[r] + bv;
        z = (gid == 2) ? fast_tanh(z) : fast_sigmoid(z);
        gout[(size_t)(rowbase + r) * G4H + ncol] = z;
    }
}

// ---------------------------------------------------------------------------
// Per-timestep recurrent GEMM, both directions in one launch.
//   z[dir] = [x_t(dir) ; h(dir)] @ Wpack(dir) + bias(dir) -> act -> gates
// Grid: 64 blocks x 256 thr = 512 waves = 2 dirs * 8 Mtiles * 32 Ntile-pairs.
// Each wave: two adjacent 16x16 C tiles, one shared A tile per 32-K chunk.
// ---------------------------------------------------------------------------
template <int K1>
__global__ void __launch_bounds__(256)
k_lstm_gemm(const bf16_t* __restrict__ xseq,   // [T][B][K1]
            const bf16_t* __restrict__ hbf_f,  // [B][H]
            const bf16_t* __restrict__ hbf_b,  // [B][H]
            const bf16_t* __restrict__ Wp_f,   // packed [64 N tiles, (K1+H)/32 K chunks]
            const bf16_t* __restrict__ Wp_b,
            const float*  __restrict__ bias_f, // [4H]
            const float*  __restrict__ bias_b,
            float* __restrict__ gates,         // [2][B][4H]
            int t) {
    constexpr int KT1 = K1 / 32;               // x-phase K chunks (4 or 16)
    constexpr int KTH = Hn / 32;               // h-phase K chunks (8)
    constexpr int KT  = KT1 + KTH;

    const int wave = threadIdx.x >> 5;
    const int lane = threadIdx.x & 31;
    const int gw   = blockIdx.x * 8 + wave;    // 0..511
    const int dir  = gw >> 8;                  // 0 = fwd, 1 = bwd
    const int tile = gw & 255;
    const int mt   = tile & 7;                 // 8 M tiles (B=128)
    const int ntg  = tile >> 3;                // 32 N tile pairs (4H=1024)
    const int nt0  = ntg * 2;
    const int nt1  = nt0 + 1;

    const int tx = dir ? (Tn - 1 - t) : t;
    const bf16_t* __restrict__ xbase = xseq + (size_t)tx * Bn * K1;
    const bf16_t* __restrict__ hbf   = dir ? hbf_b : hbf_f;
    const bf16_t* __restrict__ Wp    = dir ? Wp_b  : Wp_f;
    const float*  __restrict__ bias  = dir ? bias_b : bias_f;

    const int m    = mt * 16 + (lane & 15);        // A row for this lane
    const int aoff = (lane & 16) ? 8 : 0;          // ISA A-layout lane split

    // Per-lane base pointers; all loop offsets are compile-time immediates.
    const bf16_t* __restrict__ arow_x = xbase + (size_t)m * K1 + aoff;
    const bf16_t* __restrict__ arow_h = hbf + (size_t)m * Hn + aoff;
    const bf16_t* __restrict__ w0 = Wp + (((size_t)nt0 * KT) * 32 + lane) * 16;
    const bf16_t* __restrict__ w1 = Wp + (((size_t)nt1 * KT) * 32 + lane) * 16;

    v8f acc0 = {0.f, 0.f, 0.f, 0.f, 0.f, 0.f, 0.f, 0.f};
    v8f acc1 = {0.f, 0.f, 0.f, 0.f, 0.f, 0.f, 0.f, 0.f};

    // -------- x phase: A from x_t ------------------------------------------
    #pragma unroll 4
    for (int kt = 0; kt < KT1; ++kt) {
        union { v16bf v; v8bf h[2]; } A;
        A.h[0] = *(const v8bf*)(arow_x + kt * 32);        // K=kb+aoff..+7
        A.h[1] = *(const v8bf*)(arow_x + kt * 32 + 16);   // K=kb+aoff+16..+23
        v16bf B0 = *(const v16bf*)(w0 + kt * 512);
        v16bf B1 = *(const v16bf*)(w1 + kt * 512);
        acc0 = __builtin_amdgcn_wmma_f32_16x16x32_bf16(
                   false, A.v, false, B0, (short)0, acc0, false, false);
        acc1 = __builtin_amdgcn_wmma_f32_16x16x32_bf16(
                   false, A.v, false, B1, (short)0, acc1, false, false);
    }
    // -------- h phase: A from recurrent h ----------------------------------
    #pragma unroll
    for (int kt = 0; kt < KTH; ++kt) {
        union { v16bf v; v8bf h[2]; } A;
        A.h[0] = *(const v8bf*)(arow_h + kt * 32);
        A.h[1] = *(const v8bf*)(arow_h + kt * 32 + 16);
        v16bf B0 = *(const v16bf*)(w0 + (KT1 + kt) * 512);
        v16bf B1 = *(const v16bf*)(w1 + (KT1 + kt) * 512);
        acc0 = __builtin_amdgcn_wmma_f32_16x16x32_bf16(
                   false, A.v, false, B0, (short)0, acc0, false, false);
        acc1 = __builtin_amdgcn_wmma_f32_16x16x32_bf16(
                   false, A.v, false, B1, (short)0, acc1, false, false);
    }

    float* __restrict__ gout = gates + (size_t)dir * Bn * G4H;
    store_gate_tile(acc0, mt, nt0, lane, bias, gout);
    store_gate_tile(acc1, mt, nt1, lane, bias, gout);
}

// ---------------------------------------------------------------------------
// Pointwise LSTM cell update for both directions:
//   c = f*c + i*g ; h = o*tanh(c); write f32 state, bf16 operand, seq (opt).
// ---------------------------------------------------------------------------
__global__ void k_lstm_point(const float* __restrict__ gates,   // [2][B][4H]
                             float* __restrict__ h_f, float* __restrict__ c_f,
                             float* __restrict__ h_b, float* __restrict__ c_b,
                             bf16_t* __restrict__ hbf_f, bf16_t* __restrict__ hbf_b,
                             bf16_t* __restrict__ seq,          // [T][B][2H] or null
                             int t) {
    int idx = blockIdx.x * blockDim.x + threadIdx.x;
    if (idx >= 2 * Bn * Hn) return;
    int n   = idx & (Hn - 1);
    int b   = (idx >> 8) & (Bn - 1);
    int dir = idx >> 15;

    const float* g = gates + (size_t)dir * Bn * G4H + (size_t)b * G4H;
    float gi = g[n];
    float gf = g[n + Hn];
    float gg = g[n + 2 * Hn];
    float go = g[n + 3 * Hn];

    float*  cs = dir ? c_b : c_f;
    float*  hs = dir ? h_b : h_f;
    bf16_t* hb = dir ? hbf_b : hbf_f;

    size_t s = (size_t)b * Hn + n;
    float c = gf * cs[s] + gi * gg;
    float h = go * fast_tanh(c);
    cs[s] = c;
    hs[s] = h;
    hb[s] = (bf16_t)h;
    if (seq) {
        int ts = dir ? (Tn - 1 - t) : t;     // hs_b[::-1] concat layout
        seq[((size_t)ts * Bn + b) * (2 * Hn) + dir * Hn + n] = (bf16_t)h;
    }
}

// ---------------------------------------------------------------------------
// Dense + softmax: out[b] = softmax(concat(h2f[b], h2b[b]) @ Wd + bd)
// ---------------------------------------------------------------------------
__global__ void k_dense_softmax(const float* __restrict__ h2f,
                                const float* __restrict__ h2b,
                                const float* __restrict__ Wd,  // [2H, C]
                                const float* __restrict__ bd,  // [C]
                                float* __restrict__ out) {     // [B, C]
    int b = blockIdx.x * blockDim.x + threadIdx.x;
    if (b >= Bn) return;
    float logit[Cn];
    #pragma unroll
    for (int c = 0; c < Cn; ++c) logit[c] = bd[c];
    for (int k = 0; k < 2 * Hn; ++k) {
        float hv = (k < Hn) ? h2f[(size_t)b * Hn + k]
                            : h2b[(size_t)b * Hn + (k - Hn)];
        #pragma unroll
        for (int c = 0; c < Cn; ++c) logit[c] += hv * Wd[(size_t)k * Cn + c];
    }
    float mx = logit[0];
    #pragma unroll
    for (int c = 1; c < Cn; ++c) mx = fmaxf(mx, logit[c]);
    float sum = 0.f;
    #pragma unroll
    for (int c = 0; c < Cn; ++c) { logit[c] = __expf(logit[c] - mx); sum += logit[c]; }
    float inv = 1.0f / sum;
    #pragma unroll
    for (int c = 0; c < Cn; ++c) out[(size_t)b * Cn + c] = logit[c] * inv;
}

// ---------------------------------------------------------------------------
// Host launcher
// ---------------------------------------------------------------------------
extern "C" void kernel_launch(void* const* d_in, const int* in_sizes, int n_in,
                              void* d_out, int out_size, void* d_ws, size_t ws_size,
                              hipStream_t stream) {
    (void)in_sizes; (void)n_in; (void)out_size; (void)ws_size;

    const int*   x   = (const int*)  d_in[0];
    const float* emb = (const float*)d_in[1];
    const float* W1f = (const float*)d_in[2];
    const float* U1f = (const float*)d_in[3];
    const float* b1f = (const float*)d_in[4];
    const float* W1b = (const float*)d_in[5];
    const float* U1b = (const float*)d_in[6];
    const float* b1b = (const float*)d_in[7];
    const float* W2f = (const float*)d_in[8];
    const float* U2f = (const float*)d_in[9];
    const float* b2f = (const float*)d_in[10];
    const float* W2b = (const float*)d_in[11];
    const float* U2b = (const float*)d_in[12];
    const float* b2b = (const float*)d_in[13];
    const float* Wd  = (const float*)d_in[14];
    const float* bd  = (const float*)d_in[15];

    // ---- workspace carve (256B aligned) -----------------------------------
    char* base = (char*)d_ws;
    size_t off = 0;
    auto carve = [&](size_t bytes) -> char* {
        char* p = base + off;
        off += (bytes + 255) & ~(size_t)255;
        return p;
    };
    const size_t K1L1 = (size_t)(Dn + Hn);        // 384
    const size_t K1L2 = (size_t)(2 * Hn + Hn);    // 768
    bf16_t* Wp1f = (bf16_t*)carve(K1L1 * G4H * sizeof(bf16_t));
    bf16_t* Wp1b = (bf16_t*)carve(K1L1 * G4H * sizeof(bf16_t));
    bf16_t* Wp2f = (bf16_t*)carve(K1L2 * G4H * sizeof(bf16_t));
    bf16_t* Wp2b = (bf16_t*)carve(K1L2 * G4H * sizeof(bf16_t));
    bf16_t* e_bf = (bf16_t*)carve((size_t)Tn * Bn * Dn * sizeof(bf16_t));     // 16 MB
    bf16_t* seq  = (bf16_t*)carve((size_t)Tn * Bn * 2 * Hn * sizeof(bf16_t)); // 64 MB
    float*  gates = (float*)carve((size_t)2 * Bn * G4H * sizeof(float));      // 1 MB
    float*  stateF = (float*)carve((size_t)8 * Bn * Hn * sizeof(float));      // h/c x4
    bf16_t* stateB = (bf16_t*)carve((size_t)4 * Bn * Hn * sizeof(bf16_t));

    float* h1f = stateF + 0 * Bn * Hn;  float* c1f = stateF + 1 * Bn * Hn;
    float* h1b = stateF + 2 * Bn * Hn;  float* c1b = stateF + 3 * Bn * Hn;
    float* h2f = stateF + 4 * Bn * Hn;  float* c2f = stateF + 5 * Bn * Hn;
    float* h2b = stateF + 6 * Bn * Hn;  float* c2b = stateF + 7 * Bn * Hn;
    bf16_t* hbf1f = stateB + 0 * Bn * Hn;
    bf16_t* hbf1b = stateB + 1 * Bn * Hn;
    bf16_t* hbf2f = stateB + 2 * Bn * Hn;
    bf16_t* hbf2b = stateB + 3 * Bn * Hn;

    // ---- weight packing (bf16, WMMA B tile layout) ------------------------
    {
        int tot1 = (int)(K1L1 * G4H), tot2 = (int)(K1L2 * G4H);
        k_pack_b<<<(tot1 + 255) / 256, 256, 0, stream>>>(W1f, Dn,     U1f, Hn, Wp1f);
        k_pack_b<<<(tot1 + 255) / 256, 256, 0, stream>>>(W1b, Dn,     U1b, Hn, Wp1b);
        k_pack_b<<<(tot2 + 255) / 256, 256, 0, stream>>>(W2f, 2 * Hn, U2f, Hn, Wp2f);
        k_pack_b<<<(tot2 + 255) / 256, 256, 0, stream>>>(W2b, 2 * Hn, U2b, Hn, Wp2b);
    }

    // ---- embedding lookup -------------------------------------------------
    {
        int tot = Tn * Bn * Dn;
        k_embed<<<(tot + 255) / 256, 256, 0, stream>>>(x, emb, e_bf);
    }

    // ---- zero all recurrent state (deterministic start) -------------------
    {
        int nf = 8 * Bn * Hn;                       // f32 words
        int nb = (4 * Bn * Hn) / 2;                 // bf16 pairs as u32
        k_zero_u32<<<(nf + 255) / 256, 256, 0, stream>>>((unsigned int*)stateF, nf);
        k_zero_u32<<<(nb + 255) / 256, 256, 0, stream>>>((unsigned int*)stateB, nb);
    }

    // ---- layer 1 bidirectional scan (writes seq) --------------------------
    for (int t = 0; t < Tn; ++t) {
        k_lstm_gemm<Dn><<<64, 256, 0, stream>>>(e_bf, hbf1f, hbf1b, Wp1f, Wp1b,
                                                b1f, b1b, gates, t);
        k_lstm_point<<<(2 * Bn * Hn) / 256, 256, 0, stream>>>(
            gates, h1f, c1f, h1b, c1b, hbf1f, hbf1b, seq, t);
    }

    // ---- layer 2 bidirectional scan (keeps last states) -------------------
    for (int t = 0; t < Tn; ++t) {
        k_lstm_gemm<2 * Hn><<<64, 256, 0, stream>>>(seq, hbf2f, hbf2b, Wp2f, Wp2b,
                                                    b2f, b2b, gates, t);
        k_lstm_point<<<(2 * Bn * Hn) / 256, 256, 0, stream>>>(
            gates, h2f, c2f, h2b, c2b, hbf2f, hbf2b, (bf16_t*)nullptr, t);
    }

    // ---- dense + softmax --------------------------------------------------
    k_dense_softmax<<<1, 128, 0, stream>>>(h2f, h2b, Wd, bd, (float*)d_out);
}
